// ResidualRNNBlocks_64518998721000
// MI455X (gfx1250) — matrix operand order repaired and checked by reading, not compiled, not run on hardware
//
#include <hip/hip_runtime.h>
#include <hip/hip_bf16.h>

typedef __attribute__((ext_vector_type(16))) __bf16 v16bf;
typedef __attribute__((ext_vector_type(8)))  float  v8f;

#define HID   1024
#define BATCH 32
#define SEQ   512
#define NGATE 4096   // 4*HID
#define KDIM  1024
#define NBLK  32     // persistent blocks for recurrent kernel
#define JSL   32     // hidden units per persistent block
#define KC    128    // K chunk for recurrent A staging

// LDS layout for recurrent kernel (bytes)
#define LW_STRIDE 1032                 // 1024 + 8 bf16 pad (16B-aligned rows, spreads banks)
#define LW_BYTES  (128 * LW_STRIDE * 2)        // 264192
#define LH_STRIDE 136                  // 128 + 8 bf16 pad
#define LH_BYTES  (32 * LH_STRIDE * 2)         // 8704
#define LG_STRIDE 132                  // 128 + 4 f32 pad
#define LG_BYTES  (32 * LG_STRIDE * 4)         // 16896
#define REC_SMEM  (LW_BYTES + LH_BYTES + LG_BYTES)  // 289792 <= 320KB

__device__ __forceinline__ float fsig(float x) {
  float e = __builtin_amdgcn_exp2f(-1.4426950408889634f * x);
  return __builtin_amdgcn_rcpf(1.0f + e);
}
__device__ __forceinline__ float ftanh(float x) {
  return 2.0f * fsig(2.0f * x) - 1.0f;
}

// CDNA5 async global->LDS copy of 16 bytes per lane (ASYNCcnt-tracked).
// VDST operand = VGPR holding LDS byte offset (flat LDS addr low 32 bits).
__device__ __forceinline__ void async_ld16(void* lds, const void* g) {
  unsigned loff = (unsigned)(uintptr_t)lds;
  unsigned long long ga = (unsigned long long)(uintptr_t)g;
  asm volatile("global_load_async_to_lds_b128 %0, %1, off"
               :: "v"(loff), "v"(ga) : "memory");
}
__device__ __forceinline__ void wait_async0() {
  asm volatile("s_wait_asynccnt 0x0" ::: "memory");
}

// ---------------- fp32 -> bf16 conversion (vectorized x4) ----------------
__global__ void cvt_f32_bf16(const float* __restrict__ src,
                             __bf16* __restrict__ dst, int n4) {
  int i = blockIdx.x * blockDim.x + threadIdx.x;
  if (i >= n4) return;
  float4 f = ((const float4*)src)[i];
  union { __bf16 h[4]; uint2 u; } o;
  o.h[0] = (__bf16)f.x; o.h[1] = (__bf16)f.y;
  o.h[2] = (__bf16)f.z; o.h[3] = (__bf16)f.w;
  ((uint2*)dst)[i] = o.u;
}

// ---------------- init: zero h_{-1} slot + global barrier counter --------
__global__ void init_state(__bf16* __restrict__ hseq0, unsigned* __restrict__ ctr) {
  int i = blockIdx.x * blockDim.x + threadIdx.x;
  if (i < BATCH * HID) hseq0[i] = (__bf16)0.0f;
  if (i == 0) *ctr = 0u;
}

// ---------------- projection GEMM: C[M,4096] = A[M,1024] * W[4096,1024]^T + b1 + b2
// Block tile 256(M) x 64(N), 256 threads = 8 waves, wave w owns rows [32w,32w+32)
// (two 16-row M-tiles -> each B fragment feeds two WMMAs).
__global__ __launch_bounds__(256) void proj_gemm(
    const __bf16* __restrict__ A,   // [M, K] row-major bf16
    const __bf16* __restrict__ W,   // [N, K] row-major bf16 (acts as B^T)
    const float* __restrict__ b1, const float* __restrict__ b2,
    float* __restrict__ C) {
  __shared__ __bf16 la[256 * 40];   // A tile, row stride 40 bf16 (80B, 16B aligned)
  __shared__ __bf16 lb[64 * 40];    // B tile (W rows), same stride
  const int tid  = threadIdx.x;
  const int wv   = tid >> 5;
  const int lane = tid & 31;
  const int l16  = lane & 15;
  const int kh   = lane >> 4;       // K-half select per ISA fragment layout
  const long n0 = (long)blockIdx.x * 64;
  const long m0 = (long)blockIdx.y * 256;

  v8f acc[2][4] = {};
  for (int kc = 0; kc < KDIM; kc += 32) {
    // A tile: 256 rows x 32 bf16 = 1024 x 16B async copies, 4 per thread
#pragma unroll
    for (int r = 0; r < 4; ++r) {
      int i = tid + r * 256;
      int row = i >> 2, kq = (i & 3) * 8;
      async_ld16(&la[row * 40 + kq], &A[(m0 + row) * KDIM + kc + kq]);
    }
    // B tile: 64 W-rows x 32 bf16 = 256 copies, 1 per thread
    {
      int row = tid >> 2, kq = (tid & 3) * 8;
      async_ld16(&lb[row * 40 + kq], &W[(n0 + row) * KDIM + kc + kq]);
    }
    wait_async0();
    __syncthreads();

    // A fragments: lane holds row M=l16 (+16*mt+32*wv), K runs [kh*8,+8) and +16
    union { uint4 u[2]; v16bf v; } af[2];
#pragma unroll
    for (int mt = 0; mt < 2; ++mt) {
      const __bf16* ap = &la[(wv * 32 + mt * 16 + l16) * 40 + kh * 8];
      af[mt].u[0] = *(const uint4*)ap;
      af[mt].u[1] = *(const uint4*)(ap + 16);
    }
#pragma unroll
    for (int ns = 0; ns < 4; ++ns) {
      // B fragment: lane holds col N=l16, K runs [kh*8,+8) and +16
      union { uint4 u[2]; v16bf v; } bf_;
      const __bf16* bp = &lb[(ns * 16 + l16) * 40 + kh * 8];
      bf_.u[0] = *(const uint4*)bp;
      bf_.u[1] = *(const uint4*)(bp + 16);
#pragma unroll
      for (int mt = 0; mt < 2; ++mt) {
        acc[mt][ns] = __builtin_amdgcn_wmma_f32_16x16x32_bf16(
            false, af[mt].v, false, bf_.v, (short)0, acc[mt][ns], false, false);
      }
    }
    __syncthreads();
  }
  // Epilogue: C/D layout — lane l16 = N col, VGPR v -> M = kh*8 + v
#pragma unroll
  for (int ns = 0; ns < 4; ++ns) {
    long col = n0 + ns * 16 + l16;
    float bias = b1[col] + b2[col];
#pragma unroll
    for (int mt = 0; mt < 2; ++mt) {
#pragma unroll
      for (int v = 0; v < 8; ++v) {
        long row = m0 + wv * 32 + mt * 16 + kh * 8 + v;
        C[row * NGATE + col] = acc[mt][ns][v] + bias;
      }
    }
  }
}

// ---------------- persistent LSTM recurrence -----------------------------
// 32 blocks, block bid owns hidden slice [bid*32, bid*32+32).
// Whh slice (128 gate rows x 1024 K, bf16) lives in LDS for all 512 steps.
// xg row index = b*sB + t*sT (layer0: sB=SEQ,sT=1; layer1: sB=1,sT=BATCH).
__global__ __launch_bounds__(256) void lstm_rec(
    const float* __restrict__ xg,     // [., 4096] fp32 pre-projections
    const __bf16* __restrict__ Whh,   // [4096, 1024] bf16 row-major
    float* __restrict__ out,          // [B, T, H] fp32
    __bf16* __restrict__ hseq,        // [(T+1), B, H] bf16, slot 0 = zeros
    unsigned* __restrict__ ctr,
    int sB, int sT, int accumulate) {
  extern __shared__ char smem[];
  __bf16* lw = (__bf16*)smem;                       // weights [128][LW_STRIDE]
  __bf16* lhh = (__bf16*)(smem + LW_BYTES);         // h chunk [32][LH_STRIDE]
  float*  lg = (float*)(smem + LW_BYTES + LH_BYTES);// gates   [32][LG_STRIDE]

  const int tid  = threadIdx.x;
  const int wv   = tid >> 5;
  const int lane = tid & 31;
  const int l16  = lane & 15;
  const int kh   = lane >> 4;
  const int j0   = blockIdx.x * JSL;

  // Preload weight slice via async DMA: row n' = g*32+jj <-> Whh row g*HID+j0+jj
  for (int i = tid; i < 128 * 128; i += 256) {
    int np = i >> 7, kq = (i & 127) * 8;
    int g = np >> 5, jj = np & 31;
    async_ld16(&lw[np * LW_STRIDE + kq],
               &Whh[(long)(g * HID + j0 + jj) * KDIM + kq]);
  }
  float c[4] = {0.f, 0.f, 0.f, 0.f};   // cell state lives in registers
  wait_async0();
  __syncthreads();

  for (int t = 0; t < SEQ; ++t) {
    v8f acc0 = {}, acc1 = {};          // batches 0-15 / 16-31, 16 cols per wave
    const __bf16* hprev = hseq + (long)t * BATCH * HID;

    for (int kc = 0; kc < KDIM; kc += KC) {
      // stage h chunk: 32 rows x 128 bf16 = 512 x 16B async copies, 2 per thread
#pragma unroll
      for (int r = 0; r < 2; ++r) {
        int i = tid + r * 256;
        int row = i >> 4, kq = (i & 15) * 8;
        async_ld16(&lhh[row * LH_STRIDE + kq],
                   &hprev[(long)row * HID + kc + kq]);
      }
      wait_async0();
      __syncthreads();
#pragma unroll
      for (int ks = 0; ks < 4; ++ks) {
        union { uint4 u[2]; v16bf v; } bfr;
        const __bf16* bp = &lw[(wv * 16 + l16) * LW_STRIDE + kc + ks * 32 + kh * 8];
        bfr.u[0] = *(const uint4*)bp;
        bfr.u[1] = *(const uint4*)(bp + 16);

        union { uint4 u[2]; v16bf v; } afr;
        const __bf16* ap0 = &lhh[l16 * LH_STRIDE + ks * 32 + kh * 8];
        afr.u[0] = *(const uint4*)ap0;
        afr.u[1] = *(const uint4*)(ap0 + 16);
        acc0 = __builtin_amdgcn_wmma_f32_16x16x32_bf16(
            false, afr.v, false, bfr.v, (short)0, acc0, false, false);

        const __bf16* ap1 = &lhh[(16 + l16) * LH_STRIDE + ks * 32 + kh * 8];
        afr.u[0] = *(const uint4*)ap1;
        afr.u[1] = *(const uint4*)(ap1 + 16);
        acc1 = __builtin_amdgcn_wmma_f32_16x16x32_bf16(
            false, afr.v, false, bfr.v, (short)0, acc1, false, false);
      }
      __syncthreads();
    }

    // gates -> LDS [batch][local gate col n' = g*32+jj]
#pragma unroll
    for (int v = 0; v < 8; ++v) {
      int np = wv * 16 + l16;
      lg[(kh * 8 + v) * LG_STRIDE + np]      = acc0[v];
      lg[(16 + kh * 8 + v) * LG_STRIDE + np] = acc1[v];
    }
    __syncthreads();

    // elementwise: thread handles 4 (b, jj) pairs, c[] in registers
#pragma unroll
    for (int e = 0; e < 4; ++e) {
      int idx = tid + 256 * e;
      int b = idx >> 5, jj = idx & 31;
      long xr = (long)b * sB + (long)t * sT;
      const float* xrow = xg + xr * NGATE + j0 + jj;
      float gi = lg[b * LG_STRIDE + jj]      + xrow[0];
      float gf = lg[b * LG_STRIDE + 32 + jj] + xrow[HID];
      float gg = lg[b * LG_STRIDE + 64 + jj] + xrow[2 * HID];
      float go = lg[b * LG_STRIDE + 96 + jj] + xrow[3 * HID];
      gi = fsig(gi); gf = fsig(gf); go = fsig(go); gg = ftanh(gg);
      float cn = gf * c[e] + gi * gg;
      c[e] = cn;
      float h = go * ftanh(cn);
      long oidx = ((long)b * SEQ + t) * HID + j0 + jj;
      if (accumulate) out[oidx] += h; else out[oidx] = h;
      hseq[((long)(t + 1) * BATCH + b) * HID + j0 + jj] = (__bf16)h;
    }

    // device-wide barrier: all 32 blocks must publish h_t before step t+1
    __threadfence();
    __syncthreads();
    if (tid == 0) {
      __hip_atomic_fetch_add(ctr, 1u, __ATOMIC_RELEASE, __HIP_MEMORY_SCOPE_AGENT);
      unsigned tgt = (unsigned)NBLK * (unsigned)(t + 1);
      while (__hip_atomic_load(ctr, __ATOMIC_ACQUIRE, __HIP_MEMORY_SCOPE_AGENT) < tgt)
        __builtin_amdgcn_s_sleep(1);
    }
    __syncthreads();
    (void)__hip_atomic_load(ctr, __ATOMIC_ACQUIRE, __HIP_MEMORY_SCOPE_AGENT);
  }
}

extern "C" void kernel_launch(void* const* d_in, const int* in_sizes, int n_in,
                              void* d_out, int out_size, void* d_ws, size_t ws_size,
                              hipStream_t stream) {
  const float* x    = (const float*)d_in[0];
  const float* Wih0 = (const float*)d_in[1];
  const float* Whh0 = (const float*)d_in[2];
  const float* bih0 = (const float*)d_in[3];
  const float* bhh0 = (const float*)d_in[4];
  const float* Wih1 = (const float*)d_in[5];
  const float* Whh1 = (const float*)d_in[6];
  const float* bih1 = (const float*)d_in[7];
  const float* bhh1 = (const float*)d_in[8];
  float* out = (float*)d_out;

  // workspace partition (all sizes multiple of 256 bytes)
  char* p = (char*)d_ws;
  __bf16* xbf   = (__bf16*)p; p += (size_t)BATCH * SEQ * KDIM * 2;   // 32 MB
  __bf16* wih0b = (__bf16*)p; p += (size_t)NGATE * KDIM * 2;         // 8 MB
  __bf16* whh0b = (__bf16*)p; p += (size_t)NGATE * KDIM * 2;
  __bf16* wih1b = (__bf16*)p; p += (size_t)NGATE * KDIM * 2;
  __bf16* whh1b = (__bf16*)p; p += (size_t)NGATE * KDIM * 2;
  float*  xg    = (float*)p;  p += (size_t)BATCH * SEQ * NGATE * 4;  // 256 MB
  __bf16* hseq  = (__bf16*)p; p += (size_t)(SEQ + 1) * BATCH * HID * 2;
  unsigned* ctr = (unsigned*)p;

  (void)hipFuncSetAttribute((const void*)lstm_rec,
                            hipFuncAttributeMaxDynamicSharedMemorySize, REC_SMEM);

  // bf16 conversions
  cvt_f32_bf16<<<(BATCH * SEQ * KDIM / 4 + 255) / 256, 256, 0, stream>>>(x, xbf, BATCH * SEQ * KDIM / 4);
  cvt_f32_bf16<<<(NGATE * KDIM / 4 + 255) / 256, 256, 0, stream>>>(Wih0, wih0b, NGATE * KDIM / 4);
  cvt_f32_bf16<<<(NGATE * KDIM / 4 + 255) / 256, 256, 0, stream>>>(Whh0, whh0b, NGATE * KDIM / 4);
  cvt_f32_bf16<<<(NGATE * KDIM / 4 + 255) / 256, 256, 0, stream>>>(Wih1, wih1b, NGATE * KDIM / 4);
  cvt_f32_bf16<<<(NGATE * KDIM / 4 + 255) / 256, 256, 0, stream>>>(Whh1, whh1b, NGATE * KDIM / 4);

  dim3 pgrid(NGATE / 64, (BATCH * SEQ) / 256);

  // ---- layer 0 ----
  // xg0[b*T+t] = x @ Wih0^T + bih0 + bhh0
  proj_gemm<<<pgrid, 256, 0, stream>>>(xbf, wih0b, bih0, bhh0, xg);
  init_state<<<(BATCH * HID + 255) / 256, 256, 0, stream>>>(hseq, ctr);
  // scan: writes out (h0 fp32) and hseq (bf16, time-major, slots 1..T)
  lstm_rec<<<NBLK, 256, REC_SMEM, stream>>>(xg, whh0b, out, hseq, ctr,
                                            /*sB=*/SEQ, /*sT=*/1, /*accumulate=*/0);

  // ---- layer 1 ----
  // xg1[t*B+b] = h0 @ Wih1^T + bih1 + bhh1   (h0 is hseq slots 1..T, time-major)
  proj_gemm<<<pgrid, 256, 0, stream>>>(hseq + (size_t)BATCH * HID, wih1b, bih1, bhh1, xg);
  init_state<<<(BATCH * HID + 255) / 256, 256, 0, stream>>>(hseq, ctr);
  // scan: accumulates h1 into out (residual add); reuses hseq for its own state
  lstm_rec<<<NBLK, 256, REC_SMEM, stream>>>(xg, whh1b, out, hseq, ctr,
                                            /*sB=*/1, /*sT=*/BATCH, /*accumulate=*/1);
}
